// Attention_83580063580205
// MI455X (gfx1250) — matrix operand order repaired and checked by reading, not compiled, not run on hardware
//
#include <hip/hip_runtime.h>
#include <hip/hip_bf16.h>

#define DIM 768
#define NH 12
#define HD 64
#define NB 8
#define SEQ 1024
#define BN (NB * SEQ)   // 8192 token rows
#define BHCNT (NB * NH) // 96 (b,h) pairs
#define VPAD 40         // padded LDS row stride (elems) for transposed V tile

typedef __attribute__((ext_vector_type(16))) __bf16 v16bf;
typedef __attribute__((ext_vector_type(8)))  __bf16 v8bf;
typedef __attribute__((ext_vector_type(4)))  __bf16 v4bf;
typedef __attribute__((ext_vector_type(8)))  float  v8f;

static __device__ __forceinline__ __bf16 f2bf(float f) { return (__bf16)f; }

static __device__ __forceinline__ v8f wmma_bf16(v16bf a, v16bf b, v8f c) {
  // D = A(16x32) * B(32x16) + C, f32 accumulate
  return __builtin_amdgcn_wmma_f32_16x16x32_bf16(false, a, false, b, (short)0, c,
                                                 false, false);
}

// A-fragment: source row-major [M][K], tile origin passed in, row stride ldk.
// Lane l holds row (l&15); 16 contiguous K starting at k0 = 16*(l>>4)
// (matches load_b_frag K order).
static __device__ __forceinline__ v16bf load_a_frag(const __bf16* __restrict p16,
                                                    int ldk, int lane) {
  const __bf16* p = p16 + (lane & 15) * ldk + ((lane >> 4) << 4);
  v8bf lo = *(const v8bf*)p;
  v8bf hi = *(const v8bf*)(p + 8);
  v16bf a;
#pragma unroll
  for (int i = 0; i < 8; ++i) { a[i] = lo[i]; a[i + 8] = hi[i]; }
  return a;
}

// B-fragment: source stored as [N][K] row-major (i.e. weight[out][in]), stride ldk.
// Lane l holds col n=(l&15); 16 contiguous K starting at 16*(l>>4)
// (same contiguous K order as load_a_frag).
static __device__ __forceinline__ v16bf load_b_frag(const __bf16* __restrict p16,
                                                    int ldk, int lane) {
  const __bf16* p = p16 + (lane & 15) * ldk + ((lane >> 4) << 4);
  v8bf lo = *(const v8bf*)p;
  v8bf hi = *(const v8bf*)(p + 8);
  v16bf b;
#pragma unroll
  for (int i = 0; i < 8; ++i) { b[i] = lo[i]; b[i + 8] = hi[i]; }
  return b;
}

// Vectorized f32 -> bf16 (n must be a multiple of 4; true for all our buffers).
__global__ void k_f32_to_bf16(const float* __restrict src, __bf16* __restrict dst, int n) {
  int i = blockIdx.x * blockDim.x + threadIdx.x;
  const int stride = gridDim.x * blockDim.x;
  const int n4 = n >> 2;
  for (; i < n4; i += stride) {
    const float4 f = ((const float4*)src)[i];
    v4bf o;
    o[0] = f2bf(f.x); o[1] = f2bf(f.y); o[2] = f2bf(f.z); o[3] = f2bf(f.w);
    ((v4bf*)dst)[i] = o;
  }
}

// QKV GEMM: [8192 x 768] x W[out=2304][in=768]; one wave computes a 32x32 tile
// (2 A-frags x 2 B-frags -> 4 WMMA per K-step). Scatter into q/k f32 [B,H,N,64]
// (kept f32 for exact LN) and v bf16 [B,H,N,64].
__global__ void __launch_bounds__(128)
k_qkv_gemm(const __bf16* __restrict xb, const __bf16* __restrict wb,
           const float* __restrict bias,
           float* __restrict qf, float* __restrict kf, __bf16* __restrict vb) {
  const int lane = threadIdx.x & 31;
  const int nt = blockIdx.x * 4 + (threadIdx.x >> 5); // 0..71 (32-col tile)
  const int mt = blockIdx.y;                          // 0..255 (32-row tile)
  const int row0 = mt * 32, col0 = nt * 32;
  v8f c00 = {}, c01 = {}, c10 = {}, c11 = {};
#pragma unroll 2
  for (int kt = 0; kt < DIM / 32; ++kt) {
    const v16bf a0 = load_a_frag(xb + (size_t)(row0 +  0) * DIM + kt * 32, DIM, lane);
    const v16bf a1 = load_a_frag(xb + (size_t)(row0 + 16) * DIM + kt * 32, DIM, lane);
    const v16bf b0 = load_b_frag(wb + (size_t)(col0 +  0) * DIM + kt * 32, DIM, lane);
    const v16bf b1 = load_b_frag(wb + (size_t)(col0 + 16) * DIM + kt * 32, DIM, lane);
    c00 = wmma_bf16(a0, b0, c00);
    c01 = wmma_bf16(a0, b1, c01);
    c10 = wmma_bf16(a1, b0, c10);
    c11 = wmma_bf16(a1, b1, c11);
  }
#pragma unroll
  for (int g = 0; g < 2; ++g) {
    const int col = col0 + g * 16 + (lane & 15);
    const float bv = bias[col];
    const int part = col / DIM;      // uniform per wave (32-col tile inside one part/head)
    const int rem = col % DIM;
    const int h = rem >> 6, d = rem & 63;
#pragma unroll
    for (int mi = 0; mi < 2; ++mi) {
      const v8f& c = mi ? (g ? c11 : c10) : (g ? c01 : c00);
#pragma unroll
      for (int i = 0; i < 8; ++i) {
        const int r = row0 + mi * 16 + i + ((lane >> 4) << 3);
        const int bidx = r >> 10, n = r & 1023;
        const float val = c[i] + bv;
        const size_t idx = (((size_t)(bidx * NH + h) * SEQ + n) << 6) + d;
        if (part == 0)      qf[idx] = val;
        else if (part == 1) kf[idx] = val;
        else                vb[idx] = f2bf(val);
      }
    }
  }
}

// LayerNorm over hd=64 per (b,h,n) row for q and k, emit bf16. One wave per row.
__global__ void __launch_bounds__(256)
k_qk_norm(const float* __restrict qf, const float* __restrict kf,
          const float* __restrict gamma, const float* __restrict beta,
          __bf16* __restrict qb, __bf16* __restrict kb) {
  const int lane = threadIdx.x & 31;
  const int gw = blockIdx.x * 8 + (threadIdx.x >> 5); // global wave id
  const bool isq = gw < (BHCNT * SEQ);
  const int row = isq ? gw : gw - BHCNT * SEQ;
  const float* src = (isq ? qf : kf) + (size_t)row * HD;
  __bf16* dst = (isq ? qb : kb) + (size_t)row * HD;
  const float x0 = src[lane], x1 = src[lane + 32];
  float s = x0 + x1, sq = x0 * x0 + x1 * x1;
#pragma unroll
  for (int m = 16; m >= 1; m >>= 1) { s += __shfl_xor(s, m); sq += __shfl_xor(sq, m); }
  const float mean = s * (1.0f / 64.0f);
  const float var = fmaxf(sq * (1.0f / 64.0f) - mean * mean, 0.0f);
  const float rstd = rsqrtf(var + 1e-5f);
  dst[lane]      = f2bf((x0 - mean) * rstd * gamma[lane]      + beta[lane]);
  dst[lane + 32] = f2bf((x1 - mean) * rstd * gamma[lane + 32] + beta[lane + 32]);
}

// Flash attention. One wave owns 16 q-rows; 4 waves (block) share a transposed
// V chunk in LDS. Output written pre-transposed as bf16 [B,N,D] for proj GEMM.
__global__ void __launch_bounds__(128)
k_attn(const __bf16* __restrict qb, const __bf16* __restrict kbuf,
       const __bf16* __restrict vb, __bf16* __restrict ob) {
  __shared__ __attribute__((aligned(16))) __bf16 vt[HD * VPAD];    // V chunk, [d][key]
  __shared__ __attribute__((aligned(16))) __bf16 plds[4][16 * 32]; // per-wave P tile
  const int lane = threadIdx.x & 31;
  const int wave = threadIdx.x >> 5;
  const int bh = blockIdx.y;
  const int b = bh / NH, h = bh % NH;
  const int q0 = (blockIdx.x * 4 + wave) * 16;
  const __bf16* qp = qb   + (size_t)bh * SEQ * HD;
  const __bf16* kp = kbuf + (size_t)bh * SEQ * HD;
  const __bf16* vp = vb   + (size_t)bh * SEQ * HD;

  const v16bf aq0 = load_a_frag(qp + q0 * HD + 0,  HD, lane);
  const v16bf aq1 = load_a_frag(qp + q0 * HD + 32, HD, lane);

  v8f acc0 = {}, acc1 = {}, acc2 = {}, acc3 = {};
  float mrow[8], lrow[8];
#pragma unroll
  for (int i = 0; i < 8; ++i) { mrow[i] = -1e30f; lrow[i] = 0.0f; }

  for (int kc = 0; kc < SEQ; kc += 32) {
    // stage V chunk transposed: vt[d][key_local]
#pragma unroll
    for (int i = 0; i < 16; ++i) {
      const int e = i * 128 + threadIdx.x;
      const int key = e >> 6, d = e & 63;
      vt[d * VPAD + key] = vp[(size_t)(kc + key) * HD + d];
    }
    // scores: S = Q (16x64) * K^T; K rows [key][hd] are exactly B-frag layout
    const v16bf bk00 = load_b_frag(kp + (size_t)(kc + 0)  * HD + 0,  HD, lane);
    const v16bf bk01 = load_b_frag(kp + (size_t)(kc + 0)  * HD + 32, HD, lane);
    const v16bf bk10 = load_b_frag(kp + (size_t)(kc + 16) * HD + 0,  HD, lane);
    const v16bf bk11 = load_b_frag(kp + (size_t)(kc + 16) * HD + 32, HD, lane);
    v8f s0 = {}, s1 = {};
    s0 = wmma_bf16(aq0, bk00, s0); s0 = wmma_bf16(aq1, bk01, s0);
    s1 = wmma_bf16(aq0, bk10, s1); s1 = wmma_bf16(aq1, bk11, s1);

    __bf16* pw = plds[wave];
#pragma unroll
    for (int i = 0; i < 8; ++i) {
      const float sa = s0[i] * 0.125f, sb = s1[i] * 0.125f; // scale = hd^-0.5
      float vmax = fmaxf(sa, sb);
#pragma unroll
      for (int m = 8; m >= 1; m >>= 1) vmax = fmaxf(vmax, __shfl_xor(vmax, m));
      const float mn = fmaxf(mrow[i], vmax);
      const float alpha = __expf(mrow[i] - mn);
      const float p0 = __expf(sa - mn), p1 = __expf(sb - mn);
      float rs = p0 + p1;
#pragma unroll
      for (int m = 8; m >= 1; m >>= 1) rs += __shfl_xor(rs, m);
      lrow[i] = lrow[i] * alpha + rs;
      mrow[i] = mn;
      acc0[i] *= alpha; acc1[i] *= alpha; acc2[i] *= alpha; acc3[i] *= alpha;
      const int prow = i + ((lane >> 4) << 3); // C-layout row
      pw[prow * 32 + (lane & 15)]      = f2bf(p0);
      pw[prow * 32 + 16 + (lane & 15)] = f2bf(p1);
    }
    __syncthreads();
    // out += P (16x32) * V (32x64)
    const v16bf ap = load_a_frag(pw, 32, lane);
    const v16bf bv0 = load_b_frag(vt + 0  * VPAD * 16, VPAD, lane);
    const v16bf bv1 = load_b_frag(vt + 16 * VPAD,      VPAD, lane);
    const v16bf bv2 = load_b_frag(vt + 32 * VPAD,      VPAD, lane);
    const v16bf bv3 = load_b_frag(vt + 48 * VPAD,      VPAD, lane);
    acc0 = wmma_bf16(ap, bv0, acc0);
    acc1 = wmma_bf16(ap, bv1, acc1);
    acc2 = wmma_bf16(ap, bv2, acc2);
    acc3 = wmma_bf16(ap, bv3, acc3);
    __syncthreads();
  }

#pragma unroll
  for (int i = 0; i < 8; ++i) {
    const float inv = 1.0f / lrow[i];
    const int n = q0 + i + ((lane >> 4) << 3);
    const size_t base = ((size_t)(b * SEQ + n)) * DIM + h * HD + (lane & 15);
    ob[base + 0]  = f2bf(acc0[i] * inv);
    ob[base + 16] = f2bf(acc1[i] * inv);
    ob[base + 32] = f2bf(acc2[i] * inv);
    ob[base + 48] = f2bf(acc3[i] * inv);
  }
}

// Projection GEMM: [8192 x 768] x proj_w^T + bias -> f32 output. 32x32 per wave.
__global__ void __launch_bounds__(128)
k_proj_gemm(const __bf16* __restrict ab, const __bf16* __restrict wb,
            const float* __restrict bias, float* __restrict out) {
  const int lane = threadIdx.x & 31;
  const int nt = blockIdx.x * 4 + (threadIdx.x >> 5); // 0..23
  const int mt = blockIdx.y;                          // 0..255
  const int row0 = mt * 32, col0 = nt * 32;
  v8f c00 = {}, c01 = {}, c10 = {}, c11 = {};
#pragma unroll 2
  for (int kt = 0; kt < DIM / 32; ++kt) {
    const v16bf a0 = load_a_frag(ab + (size_t)(row0 +  0) * DIM + kt * 32, DIM, lane);
    const v16bf a1 = load_a_frag(ab + (size_t)(row0 + 16) * DIM + kt * 32, DIM, lane);
    const v16bf b0 = load_b_frag(wb + (size_t)(col0 +  0) * DIM + kt * 32, DIM, lane);
    const v16bf b1 = load_b_frag(wb + (size_t)(col0 + 16) * DIM + kt * 32, DIM, lane);
    c00 = wmma_bf16(a0, b0, c00);
    c01 = wmma_bf16(a0, b1, c01);
    c10 = wmma_bf16(a1, b0, c10);
    c11 = wmma_bf16(a1, b1, c11);
  }
#pragma unroll
  for (int g = 0; g < 2; ++g) {
    const int col = col0 + g * 16 + (lane & 15);
    const float bv = bias[col];
#pragma unroll
    for (int mi = 0; mi < 2; ++mi) {
      const v8f& c = mi ? (g ? c11 : c10) : (g ? c01 : c00);
#pragma unroll
      for (int i = 0; i < 8; ++i) {
        const int r = row0 + mi * 16 + i + ((lane >> 4) << 3);
        out[(size_t)r * DIM + col] = c[i] + bv;
      }
    }
  }
}

extern "C" void kernel_launch(void* const* d_in, const int* in_sizes, int n_in,
                              void* d_out, int out_size, void* d_ws, size_t ws_size,
                              hipStream_t stream) {
  (void)in_sizes; (void)n_in; (void)out_size; (void)ws_size;
  const float* x      = (const float*)d_in[0];
  const float* qkv_w  = (const float*)d_in[1];
  const float* qkv_b  = (const float*)d_in[2];
  const float* proj_w = (const float*)d_in[3];
  const float* proj_b = (const float*)d_in[4];
  const float* qg     = (const float*)d_in[5];
  const float* qbeta  = (const float*)d_in[6];
  float* out = (float*)d_out;

  char* p = (char*)d_ws;
  auto alloc = [&](size_t bytes) {
    char* r = p; p += (bytes + 255) & ~(size_t)255; return r;
  };
  __bf16* x_bf  = (__bf16*)alloc((size_t)BN * DIM * 2);
  __bf16* wq_bf = (__bf16*)alloc((size_t)3 * DIM * DIM * 2);
  __bf16* wp_bf = (__bf16*)alloc((size_t)DIM * DIM * 2);
  float*  q_f   = (float*) alloc((size_t)BN * DIM * 4);
  float*  k_f   = (float*) alloc((size_t)BN * DIM * 4);
  __bf16* v_bf  = (__bf16*)alloc((size_t)BN * DIM * 2);
  __bf16* q_bf  = (__bf16*)alloc((size_t)BN * DIM * 2);
  __bf16* k_bf  = (__bf16*)alloc((size_t)BN * DIM * 2);
  __bf16* a_bf  = (__bf16*)alloc((size_t)BN * DIM * 2);

  k_f32_to_bf16<<<dim3(1024), dim3(256), 0, stream>>>(x, x_bf, BN * DIM);
  k_f32_to_bf16<<<dim3(512),  dim3(256), 0, stream>>>(qkv_w, wq_bf, 3 * DIM * DIM);
  k_f32_to_bf16<<<dim3(256),  dim3(256), 0, stream>>>(proj_w, wp_bf, DIM * DIM);
  k_qkv_gemm<<<dim3(18, 256), dim3(128), 0, stream>>>(x_bf, wq_bf, qkv_b, q_f, k_f, v_bf);
  k_qk_norm<<<dim3(24576), dim3(256), 0, stream>>>(q_f, k_f, qg, qbeta, q_bf, k_bf);
  k_attn<<<dim3(16, 96), dim3(128), 0, stream>>>(q_bf, k_bf, v_bf, a_bf);
  k_proj_gemm<<<dim3(6, 256), dim3(128), 0, stream>>>(a_bf, wp_bf, proj_b, out);
}
